// Block_37967510896882
// MI455X (gfx1250) — compile-verified, matrix operand order checked by reading, not run on hardware
//
#include <hip/hip_runtime.h>
#include <math.h>

// ---------------------------------------------------------------------------
// Transformer block (B=4, T=1024, C=1024, H=64, HD=16, DFF=4096) for gfx1250.
// All matmuls run on v_wmma_f32_16x16x32_bf16 (f32 accumulate, bf16 operands).
// Cross-lane softmax reductions use DPP16 (VALU).  Global->LDS staging is
// register double-buffered: next tile's loads are in flight while the current
// tile's WMMAs execute.  Workspace requirement: ~96 MB.
// ---------------------------------------------------------------------------

constexpr int BDIM  = 4;
constexpr int TSEQ  = 1024;
constexpr int CEMB  = 1024;
constexpr int NHEAD = 64;
constexpr int HDIM  = 16;
constexpr int DFFN  = 4096;
constexpr float EPSLN = 1e-5f;

typedef __attribute__((ext_vector_type(8)))  unsigned short us8;
typedef __attribute__((ext_vector_type(16))) unsigned short v16u16;
typedef __attribute__((ext_vector_type(16))) __bf16         v16bf;
typedef __attribute__((ext_vector_type(8)))  float          v8f;

union BF16x16 { v16u16 u; unsigned short e[16]; us8 h[2]; };
union F32x8   { v8f v; float f[8]; };

static __device__ __forceinline__ unsigned short f2bf(float f) {
  unsigned int u = __float_as_uint(f);
  u += 0x7fffu + ((u >> 16) & 1u);            // round-to-nearest-even
  return (unsigned short)(u >> 16);
}
static __device__ __forceinline__ v16bf as_bf16(v16u16 u) {
  return __builtin_bit_cast(v16bf, u);
}

// 16-lane (DPP row) butterfly reductions: all-VALU, no LDS crossbar traffic.
static __device__ __forceinline__ float redmax16(float x) {
  int v;
  v = __builtin_amdgcn_update_dpp(0, __float_as_int(x), 0xB1, 0xf, 0xf, true);
  x = fmaxf(x, __int_as_float(v));
  v = __builtin_amdgcn_update_dpp(0, __float_as_int(x), 0x4E, 0xf, 0xf, true);
  x = fmaxf(x, __int_as_float(v));
  v = __builtin_amdgcn_update_dpp(0, __float_as_int(x), 0x141, 0xf, 0xf, true);
  x = fmaxf(x, __int_as_float(v));
  v = __builtin_amdgcn_update_dpp(0, __float_as_int(x), 0x140, 0xf, 0xf, true);
  x = fmaxf(x, __int_as_float(v));
  return x;
}
static __device__ __forceinline__ float redsum16(float x) {
  int v;
  v = __builtin_amdgcn_update_dpp(0, __float_as_int(x), 0xB1, 0xf, 0xf, true);
  x += __int_as_float(v);
  v = __builtin_amdgcn_update_dpp(0, __float_as_int(x), 0x4E, 0xf, 0xf, true);
  x += __int_as_float(v);
  v = __builtin_amdgcn_update_dpp(0, __float_as_int(x), 0x141, 0xf, 0xf, true);
  x += __int_as_float(v);
  v = __builtin_amdgcn_update_dpp(0, __float_as_int(x), 0x140, 0xf, 0xf, true);
  x += __int_as_float(v);
  return x;
}

// ---------------------------------------------------------------------------
// LayerNorm: f32 row -> bf16 row.  One block per row, 256 threads, C=1024.
// ---------------------------------------------------------------------------
__global__ __launch_bounds__(256)
void ln_bf16_kernel(const float* __restrict__ x, const float* __restrict__ g,
                    const float* __restrict__ be, unsigned short* __restrict__ out) {
  const int row  = blockIdx.x;
  const int tid  = threadIdx.x;
  const int lane = tid & 31;
  const int wave = tid >> 5;
  const float4 v = *(const float4*)(x + (size_t)row * CEMB + tid * 4);
  float s  = v.x + v.y + v.z + v.w;
  float sq = v.x * v.x + v.y * v.y + v.z * v.z + v.w * v.w;
  #pragma unroll
  for (int m = 16; m >= 1; m >>= 1) {
    s  += __shfl_xor(s,  m, 32);
    sq += __shfl_xor(sq, m, 32);
  }
  __shared__ float rs[8], rq[8];
  if (lane == 0) { rs[wave] = s; rq[wave] = sq; }
  __syncthreads();
  float tot = 0.f, totq = 0.f;
  #pragma unroll
  for (int i = 0; i < 8; ++i) { tot += rs[i]; totq += rq[i]; }
  const float mean = tot * (1.0f / CEMB);
  const float var  = totq * (1.0f / CEMB) - mean * mean;
  const float rstd = rsqrtf(var + EPSLN);
  const float4 gg = *(const float4*)(g  + tid * 4);
  const float4 bb = *(const float4*)(be + tid * 4);
  unsigned short* o = out + (size_t)row * CEMB + tid * 4;
  o[0] = f2bf((v.x - mean) * rstd * gg.x + bb.x);
  o[1] = f2bf((v.y - mean) * rstd * gg.y + bb.y);
  o[2] = f2bf((v.z - mean) * rstd * gg.z + bb.z);
  o[3] = f2bf((v.w - mean) * rstd * gg.w + bb.w);
}

// ---------------------------------------------------------------------------
// WMMA GEMM: out[M,N] = epi(A[M,K](bf16) @ B[K,N](f32 weights -> bf16)).
// WMODE 0: B row-major (k*N+n).  WMODE 1: head-blocked (H,C,16) QKV weights.
// EPI 0: store bf16.  EPI 1: +bias, +f32 residual, store f32.
// EPI 2: +bias, ReLU, store bf16.
// Block: 256 threads = 8 waves, 128x128 tile, BK=32; wave tile 64x32.
// LDS: As [m][k] stride 40, Bs TRANSPOSED [n][k] stride 40 -> all fragment
// reads are ds_load_b128, 16B aligned, bank-conflict-free (20-dword stride).
// Next-tile global loads are register double-buffered behind the WMMAs.
// ---------------------------------------------------------------------------
constexpr int LDK = 40;                      // padded k-stride (half-words)

template<int WMODE, int EPI>
__global__ __launch_bounds__(256)
void gemm_wmma_kernel(const unsigned short* __restrict__ A,
                      const float* __restrict__ Bw,
                      const float* __restrict__ bias,
                      const float* __restrict__ res,
                      void* __restrict__ outp,
                      int M, int N, int K) {
  __shared__ unsigned short As[128 * LDK];   // [m][k]
  __shared__ unsigned short Bs[128 * LDK];   // [n][k]  (transposed)
  const int tid  = threadIdx.x;
  const int lane = tid & 31;
  const int wave = tid >> 5;
  const int wm = wave & 1;                   // 2 waves over M
  const int wn = wave >> 1;                  // 4 waves over N
  const int bm = blockIdx.y * 128;
  const int bn = blockIdx.x * 128;

  v8f acc[4][2];
  #pragma unroll
  for (int i = 0; i < 4; ++i)
    #pragma unroll
    for (int j = 0; j < 2; ++j) acc[i][j] = {};

  const int ar = tid >> 2;                   // A rows ar, ar+64
  const int ak = (tid & 3) << 3;             // A col group (8 bf16)
  const int br = tid >> 4;                   // B k-rows br, br+16
  const int bc = (tid & 15) << 3;            // B col group (8)
  const int lr = lane & 15;                  // row/col within 16-tile
  const int kh = (lane >> 4) << 3;           // K-half select per wave32 layout

  const unsigned short* aP0 = A + (size_t)(bm + ar) * K + ak;
  const unsigned short* aP1 = A + (size_t)(bm + ar + 64) * K + ak;
  auto bsrc = [&](int k) -> const float* {
    if constexpr (WMODE == 0) {
      return Bw + (size_t)k * N + (bn + bc);
    } else {
      const int n0 = bn + bc;                // head = n0>>4, d0 = n0&15 in {0,8}
      return Bw + ((size_t)(n0 >> 4) * K + k) * 16 + (n0 & 15);
    }
  };

  // ---- prologue: first tile into registers ----
  us8    na0  = *(const us8*)(aP0);
  us8    na1  = *(const us8*)(aP1);
  float4 nb00 = *(const float4*)(bsrc(br));
  float4 nb01 = *(const float4*)(bsrc(br) + 4);
  float4 nb10 = *(const float4*)(bsrc(br + 16));
  float4 nb11 = *(const float4*)(bsrc(br + 16) + 4);

  for (int k0 = 0; k0 < K; k0 += 32) {
    // ---- commit staged registers to LDS ----
    *(us8*)(As + ar * LDK + ak) = na0;
    *(us8*)(As + (ar + 64) * LDK + ak) = na1;
    Bs[(bc + 0) * LDK + br] = f2bf(nb00.x);
    Bs[(bc + 1) * LDK + br] = f2bf(nb00.y);
    Bs[(bc + 2) * LDK + br] = f2bf(nb00.z);
    Bs[(bc + 3) * LDK + br] = f2bf(nb00.w);
    Bs[(bc + 4) * LDK + br] = f2bf(nb01.x);
    Bs[(bc + 5) * LDK + br] = f2bf(nb01.y);
    Bs[(bc + 6) * LDK + br] = f2bf(nb01.z);
    Bs[(bc + 7) * LDK + br] = f2bf(nb01.w);
    Bs[(bc + 0) * LDK + br + 16] = f2bf(nb10.x);
    Bs[(bc + 1) * LDK + br + 16] = f2bf(nb10.y);
    Bs[(bc + 2) * LDK + br + 16] = f2bf(nb10.z);
    Bs[(bc + 3) * LDK + br + 16] = f2bf(nb10.w);
    Bs[(bc + 4) * LDK + br + 16] = f2bf(nb11.x);
    Bs[(bc + 5) * LDK + br + 16] = f2bf(nb11.y);
    Bs[(bc + 6) * LDK + br + 16] = f2bf(nb11.z);
    Bs[(bc + 7) * LDK + br + 16] = f2bf(nb11.w);
    __syncthreads();

    // ---- issue next tile's global loads (overlap with WMMAs below) ----
    if (k0 + 32 < K) {
      const int kn = k0 + 32;
      na0  = *(const us8*)(aP0 + kn);
      na1  = *(const us8*)(aP1 + kn);
      const float* s0 = bsrc(kn + br);
      const float* s1 = bsrc(kn + br + 16);
      nb00 = *(const float4*)(s0);
      nb01 = *(const float4*)(s0 + 4);
      nb10 = *(const float4*)(s1);
      nb11 = *(const float4*)(s1 + 4);
      if (k0 + 64 < K) {
        __builtin_prefetch(aP0 + (k0 + 64), 0, 1);
        __builtin_prefetch(bsrc(k0 + 64 + br), 0, 1);
      }
    }

    // ---- fragments (all ds_load_b128), 8 WMMAs per wave per K-step ----
    v16bf bfr[2];
    #pragma unroll
    for (int sn = 0; sn < 2; ++sn) {
      const int c = wn * 32 + sn * 16 + lr;
      BF16x16 t;
      t.h[0] = *(const us8*)(Bs + c * LDK + kh);
      t.h[1] = *(const us8*)(Bs + c * LDK + 16 + kh);
      bfr[sn] = as_bf16(t.u);
    }
    #pragma unroll
    for (int sm = 0; sm < 4; ++sm) {
      const int r = wm * 64 + sm * 16 + lr;
      BF16x16 t;
      t.h[0] = *(const us8*)(As + r * LDK + kh);
      t.h[1] = *(const us8*)(As + r * LDK + 16 + kh);
      const v16bf afr = as_bf16(t.u);
      #pragma unroll
      for (int sn = 0; sn < 2; ++sn)
        acc[sm][sn] = __builtin_amdgcn_wmma_f32_16x16x32_bf16(
            false, afr, false, bfr[sn], (short)0, acc[sm][sn], false, false);
    }
    __syncthreads();
  }

  // ---- epilogue ----
  float* outf = (float*)outp;
  unsigned short* outb = (unsigned short*)outp;
  #pragma unroll
  for (int sm = 0; sm < 4; ++sm) {
    #pragma unroll
    for (int sn = 0; sn < 2; ++sn) {
      F32x8 t; t.v = acc[sm][sn];
      const int colg = bn + wn * 32 + sn * 16 + lr;
      const int rowb = bm + wm * 64 + sm * 16 + kh;   // C layout: VGPR i -> row i / i+8
      #pragma unroll
      for (int i = 0; i < 8; ++i) {
        const size_t idx = (size_t)(rowb + i) * N + colg;
        float v = t.f[i];
        if constexpr (EPI >= 1) v += bias[colg];
        if constexpr (EPI == 2) v = v > 0.f ? v : 0.f;
        if constexpr (EPI == 1) { outf[idx] = v + res[idx]; }
        else                    { outb[idx] = f2bf(v); }
      }
    }
  }
}

// ---------------------------------------------------------------------------
// Causal flash attention. grid = (T/128, B*H), 256 threads = 8 waves.
// Wave w owns query rows q0..q0+15; all waves walk the same key range in
// lockstep, so K and (transposed) V tiles are staged cooperatively in LDS
// once per 32-key step (register double-buffered).  Per step: two score
// WMMAs -> online softmax (DPP16 reductions) -> P staged via per-wave LDS
// into an A-fragment -> one PV WMMA.
// ---------------------------------------------------------------------------
__global__ __launch_bounds__(256)
void attn_wmma_kernel(const unsigned short* __restrict__ Q,
                      const unsigned short* __restrict__ Kk,
                      const unsigned short* __restrict__ V,
                      unsigned short* __restrict__ O) {
  __shared__ unsigned short Ps[8][16 * LDK];     // per-wave P tile [row][k]
  __shared__ unsigned short Ks[32 * 24];         // K tile [key][d], stride 24
  __shared__ unsigned short Vs[16 * LDK];        // V tile transposed [d][key]
  const int tid  = threadIdx.x;
  const int lane = tid & 31;
  const int wave = tid >> 5;
  const int lr = lane & 15;
  const int kh = (lane >> 4) << 3;
  const int bh = blockIdx.y;
  const size_t base = (size_t)(bh >> 6) * TSEQ * CEMB + (size_t)(bh & 63) * HDIM;
  const int q0 = blockIdx.x * 128 + wave * 16;

  v16bf qA;
  {
    BF16x16 t;
    t.h[0] = *(const us8*)(Q + base + (size_t)(q0 + lr) * CEMB + kh);
    #pragma unroll
    for (int j = 0; j < 8; ++j) t.e[8 + j] = 0;  // head-dim pad K=16..31
    qA = as_bf16(t.u);
  }
  F32x8 o;
  float m[8], s[8];
  #pragma unroll
  for (int i = 0; i < 8; ++i) { o.f[i] = 0.f; m[i] = -1e30f; s[i] = 0.f; }
  const float scale = 0.03125f;                  // C^-0.5 = 1/32 (ref scales by C!)
  const int kend = blockIdx.x * 128 + 128;       // uniform per block
  const v8f zero8 = {};

  const int skey = tid >> 3;                     // staging: key row 0..31
  const int sd   = (tid & 7) << 1;               // staging: d pair 0..14

  unsigned int kv = *(const unsigned int*)(Kk + base + (size_t)skey * CEMB + sd);
  unsigned int vv = *(const unsigned int*)(V  + base + (size_t)skey * CEMB + sd);

  for (int ks = 0; ks < kend; ks += 32) {
    // ---- commit staged K/V registers to LDS ----
    *(unsigned int*)(Ks + skey * 24 + sd) = kv;
    Vs[(sd + 0) * LDK + skey] = (unsigned short)(vv & 0xffffu);
    Vs[(sd + 1) * LDK + skey] = (unsigned short)(vv >> 16);
    __syncthreads();
    // ---- issue next step's K/V loads (overlap with compute) ----
    if (ks + 32 < kend) {
      kv = *(const unsigned int*)(Kk + base + (size_t)(ks + 32 + skey) * CEMB + sd);
      vv = *(const unsigned int*)(V  + base + (size_t)(ks + 32 + skey) * CEMB + sd);
    }

    // ---- scores: two 16-key WMMAs against resident Q fragment ----
    F32x8 c[2];
    #pragma unroll
    for (int sub = 0; sub < 2; ++sub) {
      BF16x16 t;
      t.h[0] = *(const us8*)(Ks + (sub * 16 + lr) * 24 + kh);
      #pragma unroll
      for (int j = 0; j < 8; ++j) t.e[8 + j] = 0;
      const v16bf kB = as_bf16(t.u);
      c[sub].v = __builtin_amdgcn_wmma_f32_16x16x32_bf16(
          false, qA, false, kB, (short)0, zero8, false, false);
    }
    // ---- online softmax over the 32 new columns ----
    #pragma unroll
    for (int i = 0; i < 8; ++i) {
      const int rowq = q0 + i + kh;
      float v0 = c[0].f[i] * scale;
      float v1 = c[1].f[i] * scale;
      if (ks + lr > rowq)      v0 = -1e30f;      // causal mask
      if (ks + 16 + lr > rowq) v1 = -1e30f;
      const float tm = redmax16(fmaxf(v0, v1));
      const float mn = fmaxf(m[i], tm);
      const float al = __expf(m[i] - mn);
      const float p0 = __expf(v0 - mn);
      const float p1 = __expf(v1 - mn);
      const float ps = redsum16(p0 + p1);
      s[i] = s[i] * al + ps;
      m[i] = mn;
      o.f[i] *= al;
      unsigned short* pr = &Ps[wave][(i + kh) * LDK];
      pr[lr]      = f2bf(p0);
      pr[16 + lr] = f2bf(p1);
    }
    __syncthreads();
    // ---- P@V accumulate ----
    BF16x16 pt;                                  // P as A-fragment (16x32)
    pt.h[0] = *(const us8*)(&Ps[wave][lr * LDK + kh]);
    pt.h[1] = *(const us8*)(&Ps[wave][lr * LDK + 16 + kh]);
    const v16bf pA = as_bf16(pt.u);
    BF16x16 vt;                                  // V as B-fragment (32 keys x 16)
    vt.h[0] = *(const us8*)(Vs + lr * LDK + kh);
    vt.h[1] = *(const us8*)(Vs + lr * LDK + 16 + kh);
    const v16bf vB = as_bf16(vt.u);
    o.v = __builtin_amdgcn_wmma_f32_16x16x32_bf16(
        false, pA, false, vB, (short)0, o.v, false, false);
    __syncthreads();
  }
  #pragma unroll
  for (int i = 0; i < 8; ++i) {
    const float inv = 1.0f / s[i];
    O[base + (size_t)(q0 + i + kh) * CEMB + lr] = f2bf(o.f[i] * inv);
  }
}

// ---------------------------------------------------------------------------
extern "C" void kernel_launch(void* const* d_in, const int* in_sizes, int n_in,
                              void* d_out, int out_size, void* d_ws, size_t ws_size,
                              hipStream_t stream) {
  (void)in_sizes; (void)n_in; (void)out_size; (void)ws_size;
  const float* x     = (const float*)d_in[0];
  const float* Wq    = (const float*)d_in[1];
  const float* Wk    = (const float*)d_in[2];
  const float* Wv    = (const float*)d_in[3];
  const float* Wproj = (const float*)d_in[4];
  const float* bproj = (const float*)d_in[5];
  const float* W1    = (const float*)d_in[6];
  const float* b1    = (const float*)d_in[7];
  const float* W2    = (const float*)d_in[8];
  const float* b2    = (const float*)d_in[9];
  const float* g1    = (const float*)d_in[10];
  const float* be1   = (const float*)d_in[11];
  const float* g2    = (const float*)d_in[12];
  const float* be2   = (const float*)d_in[13];

  const int BT = BDIM * TSEQ;                    // 4096 rows
  char* ws = (char*)d_ws;
  size_t off = 0;
  auto alloc = [&](size_t bytes) -> void* {
    void* p = ws + off;
    off += (bytes + 255) & ~(size_t)255;
    return p;
  };
  unsigned short* hb  = (unsigned short*)alloc((size_t)BT * CEMB * 2);
  unsigned short* Qb  = (unsigned short*)alloc((size_t)BT * CEMB * 2);
  unsigned short* Kb  = (unsigned short*)alloc((size_t)BT * CEMB * 2);
  unsigned short* Vb  = (unsigned short*)alloc((size_t)BT * CEMB * 2);
  unsigned short* Ob  = (unsigned short*)alloc((size_t)BT * CEMB * 2);
  float*          x1  = (float*)         alloc((size_t)BT * CEMB * 4);
  unsigned short* h2b = (unsigned short*)alloc((size_t)BT * CEMB * 2);
  unsigned short* ff1 = (unsigned short*)alloc((size_t)BT * DFFN * 2);

  const dim3 blk(256);
  const dim3 g_c(CEMB / 128, BT / 128);          // (8, 32)
  const dim3 g_f(DFFN / 128, BT / 128);          // (32, 32)
  const dim3 g_a(TSEQ / 128, BDIM * NHEAD);      // (8, 256)

  // 1) pre-LN 1
  ln_bf16_kernel<<<BT, blk, 0, stream>>>(x, g1, be1, hb);
  // 2) QKV projections (head-blocked weights)
  gemm_wmma_kernel<1, 0><<<g_c, blk, 0, stream>>>(hb, Wq, nullptr, nullptr, Qb, BT, CEMB, CEMB);
  gemm_wmma_kernel<1, 0><<<g_c, blk, 0, stream>>>(hb, Wk, nullptr, nullptr, Kb, BT, CEMB, CEMB);
  gemm_wmma_kernel<1, 0><<<g_c, blk, 0, stream>>>(hb, Wv, nullptr, nullptr, Vb, BT, CEMB, CEMB);
  // 3) causal flash attention
  attn_wmma_kernel<<<g_a, blk, 0, stream>>>(Qb, Kb, Vb, Ob);
  // 4) output projection + residual (f32)
  gemm_wmma_kernel<0, 1><<<g_c, blk, 0, stream>>>(Ob, Wproj, bproj, x, x1, BT, CEMB, CEMB);
  // 5) pre-LN 2
  ln_bf16_kernel<<<BT, blk, 0, stream>>>(x1, g2, be2, h2b);
  // 6) FFN up + ReLU (bf16)
  gemm_wmma_kernel<0, 2><<<g_f, blk, 0, stream>>>(h2b, W1, b1, nullptr, ff1, BT, DFFN, CEMB);
  // 7) FFN down + residual -> f32 output
  gemm_wmma_kernel<0, 1><<<g_c, blk, 0, stream>>>(ff1, W2, b2, x1, (float*)d_out, BT, CEMB, DFFN);
}
